// TemporalGNN_80582176407994
// MI455X (gfx1250) — matrix operand order repaired and checked by reading, not compiled, not run on hardware
//
#include <hip/hip_runtime.h>
#include <hip/hip_bf16.h>

#define NN 100000
#define NE 1600000
#define FIN 8
#define TP 18
#define UN 256
#define MT 80                 // M-tile (nodes per workgroup); 100000/80 = 1250 exact
#define MSUB 5                // 5 sub-tiles of 16 rows
#define NBLK (NN / MT)        // 1250

typedef __attribute__((ext_vector_type(16))) _Float16 v16h;
typedef __attribute__((ext_vector_type(8)))  _Float16 v8h;
typedef __attribute__((ext_vector_type(8)))  float    v8f;

// ---------------------------------------------------------------------------
// WMMA fragment helpers (contiguous k order, same in A and B, wave32)
// A (16x32 f16): lane L -> M = L&15, hs = L>>4.
//   a[0..15] = row[kt*32 + 16*hs + 0..15]
// -> two contiguous 16B LDS loads from a row-major tile.
__device__ __forceinline__ v16h load_a_frag(const _Float16* row, int hs) {
  v8h lo = *(const v8h*)(row + 16 * hs);
  v8h hi = *(const v8h*)(row + 16 * hs + 8);
  v16h a;
#pragma unroll
  for (int i = 0; i < 8; ++i) { a[i] = lo[i]; a[i + 8] = hi[i]; }
  return a;
}
// B (32x16 f16): lane L -> N = L&15, hs = L>>4; b[i] = B[kt*32 + 16*hs + i][N].
// Weights pre-transposed (WT[n][k], k contiguous) -> ONE 32B global load.
__device__ __forceinline__ v16h load_b_frag(const _Float16* wt_row_n, int kt, int hs) {
  return *(const v16h*)(wt_row_n + kt * 32 + 16 * hs);
}
#define WMMA_F16(acc, a, b) \
  acc = __builtin_amdgcn_wmma_f32_16x16x32_f16(false, (a), false, (b), (short)0, (acc), false, false)

__device__ __forceinline__ float gate_pre(float accr, int m, const float* ldsXA,
                                          const float* wc, float bs) {
  float s = accr + bs;
  const float* xa = ldsXA + m * FIN;
#pragma unroll
  for (int f = 0; f < FIN; ++f) s += xa[f] * wc[f];
  return s;
}
__device__ __forceinline__ float sigmoidf_(float x) { return 1.f / (1.f + __expf(-x)); }

// ---------------------------------------------------------------------------
// Setup kernels
__global__ void k_init_deg(float* deg) {
  int i = blockIdx.x * blockDim.x + threadIdx.x;
  if (i < NN) deg[i] = 1.0f;  // self-loop weight
}
__global__ void k_deg_edges(const long long* __restrict__ ei,
                            const float* __restrict__ ew, float* deg) {
  int e = blockIdx.x * blockDim.x + threadIdx.x;
  if (e >= NE) return;
  long long d = ei[NE + e];
  atomicAdd(&deg[d], ew[e]);
}
__global__ void k_dinv(float* dinv) {
  int i = blockIdx.x * blockDim.x + threadIdx.x;
  if (i >= NN) return;
  float d = dinv[i];
  dinv[i] = (d > 0.f) ? rsqrtf(d) : 0.f;
}
__global__ void k_xa_self(const float* __restrict__ x, const float* __restrict__ dinv,
                          float* __restrict__ XA) {
  int idx = blockIdx.x * blockDim.x + threadIdx.x;  // over NN*36 float4s
  if (idx >= NN * 36) return;
  int n = idx / 36;
  float s = dinv[n] * dinv[n];
  float4 v = ((const float4*)x)[idx];
  float4 o; o.x = v.x * s; o.y = v.y * s; o.z = v.z * s; o.w = v.w * s;
  ((float4*)XA)[idx] = o;
}
__global__ void k_edge_agg(const long long* __restrict__ ei, const float* __restrict__ ew,
                           const float* __restrict__ dinv, const float* __restrict__ x,
                           float* __restrict__ XA) {
  int idx = blockIdx.x * blockDim.x + threadIdx.x;
  if (idx >= NE * 36) return;
  int e = idx / 36, j = idx % 36;
  long long s = ei[e], d = ei[NE + e];
  float nw = dinv[s] * ew[e] * dinv[d];
  float4 v = ((const float4*)x)[(size_t)s * 36 + j];
  float* dst = XA + (size_t)d * 144 + j * 4;
  atomicAdd(dst + 0, nw * v.x);
  atomicAdd(dst + 1, nw * v.y);
  atomicAdd(dst + 2, nw * v.z);
  atomicAdd(dst + 3, nw * v.w);
}
__global__ void k_zero_state(float* Hf32, _Float16* Hf16, float* acc) {
  int i = blockIdx.x * blockDim.x + threadIdx.x;
  if (i >= NN * UN) return;
  Hf32[i] = 0.f; acc[i] = 0.f; Hf16[i] = (_Float16)0.f;
}
__global__ void k_softmax(const float* __restrict__ att, float* __restrict__ probs) {
  if (threadIdx.x == 0 && blockIdx.x == 0) {
    float mx = -1e30f;
    for (int i = 0; i < TP; ++i) mx = fmaxf(mx, att[i]);
    float e[TP], sm = 0.f;
    for (int i = 0; i < TP; ++i) { e[i] = __expf(att[i] - mx); sm += e[i]; }
    for (int i = 0; i < TP; ++i) probs[i] = e[i] / sm;
  }
}
// Combined small weights WcT[g][c][f] = sum_k Wg[f][k] * Wl_g[k][c]  (Wl top half)
__global__ void k_prep_wc(const float* Wz, const float* Wr, const float* Wh,
                          const float* Wlz, const float* Wlr, const float* Wlh,
                          float* WcT) {
  int idx = blockIdx.x * blockDim.x + threadIdx.x;  // 3*256*8
  if (idx >= 3 * UN * FIN) return;
  int g = idx / (UN * FIN), rem = idx % (UN * FIN);
  int c = rem / FIN, f = rem % FIN;
  const float* W  = (g == 0) ? Wz  : (g == 1) ? Wr  : Wh;
  const float* Wl = (g == 0) ? Wlz : (g == 1) ? Wlr : Wlh;
  float s = 0.f;
  for (int k = 0; k < UN; ++k) s += W[f * UN + k] * Wl[k * UN + c];
  WcT[(size_t)g * UN * FIN + c * FIN + f] = s;
}
// biasP[g][c] = bl_g[c] + sum_k b_g[k] * Wl_g[k][c]
__global__ void k_prep_bias(const float* bz, const float* br, const float* bh,
                            const float* blz, const float* blr, const float* blh,
                            const float* Wlz, const float* Wlr, const float* Wlh,
                            float* biasP) {
  int g = blockIdx.x, c = threadIdx.x;
  const float* b  = (g == 0) ? bz  : (g == 1) ? br  : bh;
  const float* bl = (g == 0) ? blz : (g == 1) ? blr : blh;
  const float* Wl = (g == 0) ? Wlz : (g == 1) ? Wlr : Wlh;
  float s = bl[c];
  for (int k = 0; k < UN; ++k) s += b[k] * Wl[k * UN + c];
  biasP[g * UN + c] = s;
}
// Generic f32 -> transposed f16 weight prep: dst[c][k] = src[(row_off+k)*ld + c]
__global__ void k_transpose_f16(const float* __restrict__ src, _Float16* __restrict__ dst,
                                int K, int C, int src_ld, int row_off, int col_valid) {
  int idx = blockIdx.x * blockDim.x + threadIdx.x;
  if (idx >= K * C) return;
  int c = idx / K, k = idx % K;
  float v = (c < col_valid) ? src[(size_t)(row_off + k) * src_ld + c] : 0.f;
  dst[(size_t)c * K + k] = (_Float16)v;
}
__global__ void k_convert_relu(const float* __restrict__ acc, _Float16* __restrict__ out) {
  int i = blockIdx.x * blockDim.x + threadIdx.x;
  if (i >= NN * UN) return;
  out[i] = (_Float16)fmaxf(acc[i], 0.f);
}

// ---------------------------------------------------------------------------
// Shared tile loaders (MT x 256 f16 tile = MT*512 bytes, as uint4 chunks)
__device__ __forceinline__ void stage_tile(const _Float16* g, _Float16* lds, int nthr) {
  const uint4* src = (const uint4*)g;
  uint4* dl = (uint4*)lds;
  for (int i = threadIdx.x; i < MT * UN / 8; i += nthr) dl[i] = src[i];
}
__device__ __forceinline__ void stage_xa(const float* XA, float* ldsXA, int nb, int t,
                                         int nthr) {
  for (int i = threadIdx.x; i < MT * FIN; i += nthr) {
    int m = i >> 3, f = i & 7;
    ldsXA[i] = XA[(size_t)(nb * MT + m) * 144 + f * TP + t];
  }
}

// Z & R gates: pre = H @ Wl_bot + XA_t @ Wc + bias'.  MT-node tile per WG;
// 32 (gate,Ntile) pairs over 8 waves -> 4 passes, each pass: one B fragment
// per K-step reused across 5 M-subtiles (5 WMMAs per B load).
__global__ void __launch_bounds__(256, 2)
k_gate_zr(const _Float16* __restrict__ Hf16, const float* __restrict__ XA, int t,
          const _Float16* __restrict__ WTz, const _Float16* __restrict__ WTr,
          const float* __restrict__ WcTz, const float* __restrict__ WcTr,
          const float* __restrict__ biasZ, const float* __restrict__ biasR,
          _Float16* __restrict__ Z16, _Float16* __restrict__ HR16) {
  __shared__ __align__(16) _Float16 ldsH[MT * UN];
  __shared__ float ldsXA[MT * FIN];
  int nb = blockIdx.x;
  stage_tile(Hf16 + (size_t)nb * MT * UN, ldsH, 256);
  stage_xa(XA, ldsXA, nb, t, 256);
  __syncthreads();

  int wv = threadIdx.x >> 5, lane = threadIdx.x & 31;
  int hs = lane >> 4, nc = lane & 15, mrow = lane & 15;

  for (int p = 0; p < 4; ++p) {
    int pair = wv + p * 8;         // 0..31 (wave-uniform)
    int gate = pair >> 4;          // 0 = Z, 1 = R
    int nt = pair & 15;
    const _Float16* bp = (gate ? WTr : WTz) + (size_t)(nt * 16 + nc) * UN;
    v8f acc[MSUB] = {};
#pragma unroll
    for (int kt = 0; kt < 8; ++kt) {
      v16h b = load_b_frag(bp, kt, hs);
#pragma unroll
      for (int ms = 0; ms < MSUB; ++ms) {
        v16h a = load_a_frag(&ldsH[(ms * 16 + mrow) * UN + kt * 32], hs);
        WMMA_F16(acc[ms], a, b);
      }
    }
    int col = nt * 16 + nc;
    const float* wc = (gate ? WcTr : WcTz) + col * FIN;
    float bs = (gate ? biasR : biasZ)[col];
    if (gate == 0) {
#pragma unroll
      for (int ms = 0; ms < MSUB; ++ms)
#pragma unroll
        for (int r = 0; r < 8; ++r) {
          int m = ms * 16 + r + 8 * hs;
          float s = gate_pre(acc[ms][r], m, ldsXA, wc, bs);
          Z16[(size_t)(nb * MT + m) * UN + col] = (_Float16)sigmoidf_(s);
        }
    } else {
#pragma unroll
      for (int ms = 0; ms < MSUB; ++ms)
#pragma unroll
        for (int r = 0; r < 8; ++r) {
          int m = ms * 16 + r + 8 * hs;
          float s = gate_pre(acc[ms][r], m, ldsXA, wc, bs);
          float hv = (float)ldsH[m * UN + col];
          HR16[(size_t)(nb * MT + m) * UN + col] = (_Float16)(hv * sigmoidf_(s));
        }
    }
  }
}

// Candidate gate + GRU update + attention accumulation. 16 Ntiles over 8
// waves -> 2 passes x 5 M-subtiles.
__global__ void __launch_bounds__(256, 2)
k_gate_h(const _Float16* __restrict__ HR16, const float* __restrict__ XA, int t,
         const _Float16* __restrict__ WTh, const float* __restrict__ WcTh,
         const float* __restrict__ biasH, const _Float16* __restrict__ Z16,
         float* __restrict__ Hf32, _Float16* __restrict__ Hf16,
         float* __restrict__ accum, const float* __restrict__ probs) {
  __shared__ __align__(16) _Float16 ldsA[MT * UN];
  __shared__ float ldsXA[MT * FIN];
  int nb = blockIdx.x;
  stage_tile(HR16 + (size_t)nb * MT * UN, ldsA, 256);
  stage_xa(XA, ldsXA, nb, t, 256);
  __syncthreads();

  int wv = threadIdx.x >> 5, lane = threadIdx.x & 31;
  int hs = lane >> 4, nc = lane & 15, mrow = lane & 15;
  float p = probs[t];

  for (int ps = 0; ps < 2; ++ps) {
    int nt = wv + ps * 8;
    const _Float16* bp = WTh + (size_t)(nt * 16 + nc) * UN;
    v8f acc[MSUB] = {};
#pragma unroll
    for (int kt = 0; kt < 8; ++kt) {
      v16h b = load_b_frag(bp, kt, hs);
#pragma unroll
      for (int ms = 0; ms < MSUB; ++ms) {
        v16h a = load_a_frag(&ldsA[(ms * 16 + mrow) * UN + kt * 32], hs);
        WMMA_F16(acc[ms], a, b);
      }
    }
    int col = nt * 16 + nc;
    const float* wc = WcTh + col * FIN;
    float bs = biasH[col];
#pragma unroll
    for (int ms = 0; ms < MSUB; ++ms)
#pragma unroll
      for (int r = 0; r < 8; ++r) {
        int m = ms * 16 + r + 8 * hs;
        size_t idx = (size_t)(nb * MT + m) * UN + col;
        float ht = tanhf(gate_pre(acc[ms][r], m, ldsXA, wc, bs));
        float z = (float)Z16[idx];
        float hn = z * Hf32[idx] + (1.f - z) * ht;
        Hf32[idx] = hn;
        Hf16[idx] = (_Float16)hn;
        accum[idx] += p * hn;
      }
  }
}

// MLP layer: out_f16 = relu(A_f16 @ W + b).  Same 2-pass x 5-subtile scheme.
__global__ void __launch_bounds__(256, 2)
k_mlp(const _Float16* __restrict__ Ain, const _Float16* __restrict__ WT,
      const float* __restrict__ bias, _Float16* __restrict__ Aout) {
  __shared__ __align__(16) _Float16 ldsA[MT * UN];
  int nb = blockIdx.x;
  stage_tile(Ain + (size_t)nb * MT * UN, ldsA, 256);
  __syncthreads();
  int wv = threadIdx.x >> 5, lane = threadIdx.x & 31;
  int hs = lane >> 4, nc = lane & 15, mrow = lane & 15;

  for (int ps = 0; ps < 2; ++ps) {
    int nt = wv + ps * 8;
    const _Float16* bp = WT + (size_t)(nt * 16 + nc) * UN;
    v8f acc[MSUB] = {};
#pragma unroll
    for (int kt = 0; kt < 8; ++kt) {
      v16h b = load_b_frag(bp, kt, hs);
#pragma unroll
      for (int ms = 0; ms < MSUB; ++ms) {
        v16h a = load_a_frag(&ldsA[(ms * 16 + mrow) * UN + kt * 32], hs);
        WMMA_F16(acc[ms], a, b);
      }
    }
    int col = nt * 16 + nc;
    float bs = bias[col];
#pragma unroll
    for (int ms = 0; ms < MSUB; ++ms)
#pragma unroll
      for (int r = 0; r < 8; ++r) {
        int m = ms * 16 + r + 8 * hs;
        Aout[(size_t)(nb * MT + m) * UN + col] = (_Float16)fmaxf(acc[ms][r] + bs, 0.f);
      }
  }
}

// Output layer: [N,256] @ [256,18] (padded to 32 cols), f32 out. 2 waves/WG,
// each wave one N-tile x 5 M-subtiles.
__global__ void __launch_bounds__(64, 8)
k_outproj(const _Float16* __restrict__ Ain, const _Float16* __restrict__ WoT,
          const float* __restrict__ bo, float* __restrict__ out) {
  __shared__ __align__(16) _Float16 ldsA[MT * UN];
  int nb = blockIdx.x;
  stage_tile(Ain + (size_t)nb * MT * UN, ldsA, 64);
  __syncthreads();
  int wv = threadIdx.x >> 5, lane = threadIdx.x & 31;
  int hs = lane >> 4, nc = lane & 15, mrow = lane & 15;
  int nt = wv;
  const _Float16* bp = WoT + (size_t)(nt * 16 + nc) * UN;
  v8f acc[MSUB] = {};
#pragma unroll
  for (int kt = 0; kt < 8; ++kt) {
    v16h b = load_b_frag(bp, kt, hs);
#pragma unroll
    for (int ms = 0; ms < MSUB; ++ms) {
      v16h a = load_a_frag(&ldsA[(ms * 16 + mrow) * UN + kt * 32], hs);
      WMMA_F16(acc[ms], a, b);
    }
  }
  int col = nt * 16 + nc;
  if (col < TP) {
    float bs = bo[col];
#pragma unroll
    for (int ms = 0; ms < MSUB; ++ms)
#pragma unroll
      for (int r = 0; r < 8; ++r) {
        int m = ms * 16 + r + 8 * hs;
        out[(size_t)(nb * MT + m) * TP + col] = acc[ms][r] + bs;
      }
  }
}

// ---------------------------------------------------------------------------
extern "C" void kernel_launch(void* const* d_in, const int* in_sizes, int n_in,
                              void* d_out, int out_size, void* d_ws, size_t ws_size,
                              hipStream_t stream) {
  const float*     x   = (const float*)d_in[0];
  const long long* ei  = (const long long*)d_in[1];
  const float*     ew  = (const float*)d_in[2];
  const float* Wz = (const float*)d_in[3],  *bz = (const float*)d_in[4];
  const float* Wlz= (const float*)d_in[5],  *blz= (const float*)d_in[6];
  const float* Wr = (const float*)d_in[7],  *br = (const float*)d_in[8];
  const float* Wlr= (const float*)d_in[9],  *blr= (const float*)d_in[10];
  const float* Wh = (const float*)d_in[11], *bh = (const float*)d_in[12];
  const float* Wlh= (const float*)d_in[13], *blh= (const float*)d_in[14];
  const float* att= (const float*)d_in[15];
  const float* W1 = (const float*)d_in[16], *b1 = (const float*)d_in[17];
  const float* W2 = (const float*)d_in[18], *b2 = (const float*)d_in[19];
  const float* W3 = (const float*)d_in[20], *b3 = (const float*)d_in[21];
  const float* Wo = (const float*)d_in[22], *bo = (const float*)d_in[23];
  float* out = (float*)d_out;

  char* ws = (char*)d_ws;
  size_t off = 0;
  auto alloc = [&](size_t bytes) { size_t r = off; off += (bytes + 255) & ~(size_t)255; return r; };
  float*     dinv = (float*)(ws + alloc((size_t)NN * 4));
  float*     XAb  = (float*)(ws + alloc((size_t)NN * 144 * 4));
  float*     Hf32 = (float*)(ws + alloc((size_t)NN * UN * 4));
  _Float16*  Hf16 = (_Float16*)(ws + alloc((size_t)NN * UN * 2));
  _Float16*  HR16 = (_Float16*)(ws + alloc((size_t)NN * UN * 2));
  _Float16*  Z16  = (_Float16*)(ws + alloc((size_t)NN * UN * 2));
  float*     accb = (float*)(ws + alloc((size_t)NN * UN * 4));
  float*     probs= (float*)(ws + alloc(256));
  float*     WcT  = (float*)(ws + alloc((size_t)3 * UN * FIN * 4));
  float*     biasP= (float*)(ws + alloc((size_t)3 * UN * 4));
  _Float16*  WTg  = (_Float16*)(ws + alloc((size_t)3 * UN * UN * 2));
  _Float16*  WTm  = (_Float16*)(ws + alloc((size_t)3 * UN * UN * 2));
  _Float16*  WoT  = (_Float16*)(ws + alloc((size_t)32 * UN * 2));
  // MLP ping-pong reuses dead regions (XA and Hf16 dead after recurrence).
  _Float16* M0 = (_Float16*)XAb;
  _Float16* M1 = Hf16;

  // ---- graph norm + one-time aggregation XA = Ahat @ X (all timesteps) ----
  k_init_deg<<<(NN + 255) / 256, 256, 0, stream>>>(dinv);
  k_deg_edges<<<(NE + 255) / 256, 256, 0, stream>>>(ei, ew, dinv);
  k_dinv<<<(NN + 255) / 256, 256, 0, stream>>>(dinv);
  k_xa_self<<<(NN * 36 + 255) / 256, 256, 0, stream>>>(x, dinv, XAb);
  k_edge_agg<<<(NE * 36 + 255) / 256, 256, 0, stream>>>(ei, ew, dinv, x, XAb);
  k_zero_state<<<(NN * UN + 255) / 256, 256, 0, stream>>>(Hf32, Hf16, accb);
  k_softmax<<<1, 32, 0, stream>>>(att, probs);

  // ---- weight prep: combined K=8 weights, folded biases, f16 transposes ----
  k_prep_wc<<<(3 * UN * FIN + 255) / 256, 256, 0, stream>>>(Wz, Wr, Wh, Wlz, Wlr, Wlh, WcT);
  k_prep_bias<<<3, UN, 0, stream>>>(bz, br, bh, blz, blr, blh, Wlz, Wlr, Wlh, biasP);
  k_transpose_f16<<<(UN * UN + 255) / 256, 256, 0, stream>>>(Wlz, WTg + 0 * UN * UN, UN, UN, UN, UN, UN);
  k_transpose_f16<<<(UN * UN + 255) / 256, 256, 0, stream>>>(Wlr, WTg + 1 * UN * UN, UN, UN, UN, UN, UN);
  k_transpose_f16<<<(UN * UN + 255) / 256, 256, 0, stream>>>(Wlh, WTg + 2 * UN * UN, UN, UN, UN, UN, UN);
  k_transpose_f16<<<(UN * UN + 255) / 256, 256, 0, stream>>>(W1, WTm + 0 * UN * UN, UN, UN, UN, 0, UN);
  k_transpose_f16<<<(UN * UN + 255) / 256, 256, 0, stream>>>(W2, WTm + 1 * UN * UN, UN, UN, UN, 0, UN);
  k_transpose_f16<<<(UN * UN + 255) / 256, 256, 0, stream>>>(W3, WTm + 2 * UN * UN, UN, UN, UN, 0, UN);
  k_transpose_f16<<<(32 * UN + 255) / 256, 256, 0, stream>>>(Wo, WoT, UN, 32, TP, 0, TP);

  // ---- GRU recurrence: 18 steps x (ZR kernel, H kernel), WMMA GEMMs ----
  for (int t = 0; t < TP; ++t) {
    k_gate_zr<<<NBLK, 256, 0, stream>>>(Hf16, XAb, t,
        WTg + 0 * UN * UN, WTg + 1 * UN * UN,
        WcT + 0 * UN * FIN, WcT + 1 * UN * FIN,
        biasP + 0 * UN, biasP + 1 * UN, Z16, HR16);
    k_gate_h<<<NBLK, 256, 0, stream>>>(HR16, XAb, t,
        WTg + 2 * UN * UN, WcT + 2 * UN * FIN, biasP + 2 * UN,
        Z16, Hf32, Hf16, accb, probs);
  }

  // ---- MLP head ----
  k_convert_relu<<<(NN * UN + 255) / 256, 256, 0, stream>>>(accb, M0);
  k_mlp<<<NBLK, 256, 0, stream>>>(M0, WTm + 0 * UN * UN, b1, M1);
  k_mlp<<<NBLK, 256, 0, stream>>>(M1, WTm + 1 * UN * UN, b2, M0);
  k_mlp<<<NBLK, 256, 0, stream>>>(M0, WTm + 2 * UN * UN, b3, M1);
  k_outproj<<<NBLK, 64, 0, stream>>>(M1, WoT, bo, out);
}
